// BreakthroughSNN_50319836840752
// MI455X (gfx1250) — matrix operand order repaired and checked by reading, not compiled, not run on hardware
//
#include <hip/hip_runtime.h>
#include <stdint.h>

// ---------------------------------------------------------------- types ----
typedef __attribute__((ext_vector_type(16))) __bf16   v16bf;
typedef __attribute__((ext_vector_type(8)))  float    v8f;
typedef __attribute__((ext_vector_type(4)))  unsigned v4u;
typedef __attribute__((ext_vector_type(4)))  unsigned u32x4;
typedef __attribute__((ext_vector_type(8)))  int      i32x8;
typedef __attribute__((ext_vector_type(4)))  int      i32x4;

union ABu { v4u u[2]; v16bf h; };

// ------------------------------------------------------------- constants ---
#define VSZ   32000
#define DSZ   1024
#define DS2   512
#define LLAY  4
#define SEQ   256
#define BSZ   8
#define MROW  2048          /* B*SEQ */
#define NWG_REC 16
#define WAVES_REC (NWG_REC*8)

__device__ __forceinline__ unsigned short f2bf(float f) {
  unsigned u = __builtin_bit_cast(unsigned, f);
  return (unsigned short)((u + 0x7fffu + ((u >> 16) & 1u)) >> 16);
}

// ------------------------------------------------------ 16x16 WMMA helper --
// A row-major [16][K] bf16 ; Wt row-major [N][K] bf16 (i.e. B column n = Wt[n][:])
// A frag (16x32, this code's contiguous k order): lane L -> m=L&15 ; elems K=kbase..kbase+15,
// kbase=(L>>4)*16.
// B frag (32x16, same contiguous k order as A): lane L -> n=L&15 ; elems K=kbase..kbase+15,
// kbase=(L>>4)*16.
__device__ __forceinline__ v8f gemm16(const unsigned short* Arow,
                                      const unsigned short* Brow, int K) {
  v8f acc = {0.f,0.f,0.f,0.f,0.f,0.f,0.f,0.f};
  for (int kb = 0; kb < K; kb += 32) {
    ABu a, b;
    a.u[0] = *(const v4u*)(Arow + kb);
    a.u[1] = *(const v4u*)(Arow + kb + 8);
    b.u[0] = *(const v4u*)(Brow + kb);
    b.u[1] = *(const v4u*)(Brow + kb + 8);
    acc = __builtin_amdgcn_wmma_f32_16x16x32_bf16(false, a.h, false, b.h,
                                                  (short)0, acc, false, false);
  }
  return acc;
}

// --------------------------------------------------------- embed kernel ----
__global__ __launch_bounds__(256)
void embed_kernel(const int* __restrict__ ids, const float* __restrict__ tok,
                  const float* __restrict__ pos, unsigned short* __restrict__ xbf) {
  int s = blockIdx.x;
  for (int m = 0; m < 16; ++m) {
    unsigned short* dst = xbf + ((size_t)s * 16 + m) * DSZ;
    if (m < BSZ) {
      int id = ids[m * SEQ + s];
      const float* tr = tok + (size_t)id * DSZ;
      const float* pr = pos + (size_t)s * DSZ;
      for (int d = threadIdx.x; d < DSZ; d += 256) dst[d] = f2bf(tr[d] + pr[d]);
    } else {
      for (int d = threadIdx.x; d < DSZ; d += 256) dst[d] = 0;
    }
  }
}

// -------------------------------------- f32 [K][N] -> bf16 [N][K] transpose
__global__ __launch_bounds__(256)
void transpose_bf16(const float* __restrict__ src, unsigned short* __restrict__ dst,
                    int K, int N) {
  __shared__ float tile[32][33];
  int n = blockIdx.x * 32 + threadIdx.x;
  int k0 = blockIdx.y * 32;
  for (int i = threadIdx.y; i < 32; i += 8)
    tile[i][threadIdx.x] = src[(size_t)(k0 + i) * N + n];
  __syncthreads();
  int k = k0 + threadIdx.x;
  int n0 = blockIdx.x * 32;
  for (int i = threadIdx.y; i < 32; i += 8)
    dst[(size_t)(n0 + i) * K + k] = f2bf(tile[threadIdx.x][i]);
}

// ------------------------------------------------------ global barrier -----
__device__ __forceinline__ void gbar(unsigned* bar) {
  __threadfence();
  __syncthreads();
  if (threadIdx.x == 0) {
    volatile unsigned* vgen = bar + 1;
    unsigned g = *vgen;
    if (atomicAdd(bar, 1u) == (unsigned)(NWG_REC - 1)) {
      bar[0] = 0;
      __threadfence();
      atomicAdd((unsigned*)(bar + 1), 1u);
    } else {
      while (*vgen == g) __builtin_amdgcn_s_sleep(2);
    }
    __threadfence();
  }
  __syncthreads();
}

// ---------------------------------------- GEMM (M=16 padded) + LIF epilogue
__device__ void gemm_lif(const unsigned short* A, const unsigned short* Wt,
                         int K, int N,
                         const float* bias, const float* tau,
                         const float* bth, const float* ath,
                         float* mem, float* spike_f, unsigned short* spike_bf,
                         float* hid, int gw, int lane) {
  const int ntiles = N >> 4;
  const unsigned short* Arow = A + (size_t)(lane & 15) * K + (lane >> 4) * 16;
  for (int tile = gw; tile < ntiles; tile += WAVES_REC) {
    int n = tile * 16 + (lane & 15);
    const unsigned short* Brow = Wt + (size_t)n * K + (lane >> 4) * 16;
    v8f acc = gemm16(Arow, Brow, K);
    if (lane < 16) {                      // lanes 0-15 hold rows 0..7 (real B)
      float dec = __expf(-1.f / fmaxf(tau[n], 1.f));
      float th  = bth[n] + ath[n];
      float bv  = bias[n];
      for (int r = 0; r < 8; ++r) {
        int idx = r * N + n;
        float m  = mem[idx] * dec + acc[r] + bv;
        float sp = (m - th) >= 0.f ? 1.f : 0.f;
        mem[idx] = m * (1.f - sp);
        if (spike_f)  spike_f[idx]  = sp;
        if (spike_bf) spike_bf[idx] = f2bf(sp);
        if (hid)      hid[idx]      = sp;
      }
    }
  }
}

// ------------------------------------------- row LayerNorm (rows 0..7) -----
// val = (a ? a[i] : 0) + sgn * p[i]; out = LN(val)*g + b  -> fout/bfout
__device__ void ln_row(const float* a, float sgn, const float* p,
                       const float* g, const float* bt,
                       float* fout, unsigned short* bfout, int W,
                       float* s1, float* s2) {
  int row = blockIdx.x;
  int tid = threadIdx.x;
  int per = W >> 8;
  float vals[4];
  float sum = 0.f, sq = 0.f;
  for (int i = 0; i < per; ++i) {
    int idx = tid + (i << 8);
    float v = (a ? a[row * W + idx] : 0.f) + sgn * p[row * W + idx];
    vals[i] = v; sum += v; sq += v * v;
  }
  s1[tid] = sum; s2[tid] = sq;
  __syncthreads();
  for (int off = 128; off > 0; off >>= 1) {
    if (tid < off) { s1[tid] += s1[tid + off]; s2[tid] += s2[tid + off]; }
    __syncthreads();
  }
  float mu = s1[0] / (float)W;
  float var = s2[0] / (float)W - mu * mu;
  float rs = rsqrtf(var + 1e-5f);
  for (int i = 0; i < per; ++i) {
    int idx = tid + (i << 8);
    float y = (vals[i] - mu) * rs * g[idx] + bt[idx];
    if (fout)  fout[row * W + idx]  = y;
    if (bfout) bfout[row * W + idx] = f2bf(y);
  }
  __syncthreads();
}

// ------------------------------------------- persistent recurrent kernel ---
__global__ __launch_bounds__(256)
void recurrent_kernel(const unsigned short* xbf,
                      const unsigned short* wt_enc, const unsigned short* wt_gen,
                      const unsigned short* wt_inf,
                      const float* enc_b, const float* enc_tau,
                      const float* enc_bth, const float* enc_ath,
                      const float* gen_b, const float* gen_tau,
                      const float* gen_bth, const float* gen_ath,
                      const float* inf_b, const float* inf_tau,
                      const float* inf_bth, const float* inf_ath,
                      const float* ne_g, const float* ne_b,
                      const float* ns_g, const float* ns_b,
                      float* enc_mem, float* gen_mem, float* inf_mem,
                      float* states, unsigned short* states_bf,
                      float* bu, unsigned short* bu_bf,
                      float* pred, float* upd, float* hidden, unsigned* bar) {
  __shared__ float s1[256], s2[256];
  int lane = threadIdx.x & 31;
  int gw = blockIdx.x * 8 + (threadIdx.x >> 5);

  for (int t = 0; t < SEQ; ++t) {
    // encoder: bu = LIF(xt @ enc_W + enc_b)
    gemm_lif(xbf + (size_t)t * 16 * DSZ, wt_enc, DSZ, DSZ,
             enc_b, enc_tau, enc_bth, enc_ath, enc_mem, bu, bu_bf, nullptr,
             gw, lane);
    gbar(bar);

    for (int j = 0; j < LLAY; ++j) {
      // pred = LIF(states[j] @ gen_W[j] + gen_b[j])
      gemm_lif(states_bf + (size_t)j * 16 * DS2, wt_gen + (size_t)j * DSZ * DS2,
               DS2, DSZ, gen_b + j * DSZ, gen_tau + j * DSZ,
               gen_bth + j * DSZ, gen_ath + j * DSZ,
               gen_mem + (size_t)j * 8 * DSZ, pred, nullptr, nullptr, gw, lane);
      gbar(bar);
      // bu = err = LN(bu - pred)
      if (blockIdx.x < 8)
        ln_row(bu, -1.f, pred, ne_g + j * DSZ, ne_b + j * DSZ,
               bu, bu_bf, DSZ, s1, s2);
      gbar(bar);
      // upd = LIF(err @ inf_W[j] + inf_b[j])
      gemm_lif(bu_bf, wt_inf + (size_t)j * DS2 * DSZ, DSZ, DS2,
               inf_b + j * DS2, inf_tau + j * DS2, inf_bth + j * DS2,
               inf_ath + j * DS2, inf_mem + (size_t)j * 8 * DS2,
               upd, nullptr, nullptr, gw, lane);
      gbar(bar);
      // states[j] = LN(states[j] + upd)
      if (blockIdx.x < 8)
        ln_row(states + (size_t)j * 8 * DS2, 1.f, upd,
               ns_g + j * DS2, ns_b + j * DS2,
               states + (size_t)j * 8 * DS2,
               states_bf + (size_t)j * 16 * DS2, DS2, s1, s2);
      gbar(bar);
    }

    // extra pass through last PC layer, zero bottom-up input
    gemm_lif(states_bf + (size_t)3 * 16 * DS2, wt_gen + (size_t)3 * DSZ * DS2,
             DS2, DSZ, gen_b + 3 * DSZ, gen_tau + 3 * DSZ,
             gen_bth + 3 * DSZ, gen_ath + 3 * DSZ,
             gen_mem + (size_t)3 * 8 * DSZ, pred, nullptr,
             hidden + (size_t)t * 8 * DSZ, gw, lane);
    gbar(bar);
    if (blockIdx.x < 8)
      ln_row(nullptr, -1.f, pred, ne_g + 3 * DSZ, ne_b + 3 * DSZ,
             bu, bu_bf, DSZ, s1, s2);
    gbar(bar);
    gemm_lif(bu_bf, wt_inf + (size_t)3 * DS2 * DSZ, DSZ, DS2,
             inf_b + 3 * DS2, inf_tau + 3 * DS2, inf_bth + 3 * DS2,
             inf_ath + 3 * DS2, inf_mem + (size_t)3 * 8 * DS2,
             upd, nullptr, nullptr, gw, lane);
    gbar(bar);
  }
}

// --------------------------------- LN + exact GELU + bf16 convert (h) ------
__global__ __launch_bounds__(256)
void postln_gelu(const float* __restrict__ hidden, const float* __restrict__ g,
                 const float* __restrict__ bt, unsigned short* __restrict__ hb) {
  __shared__ float s1[256], s2[256];
  int row = blockIdx.x;              // row = b*SEQ + s  (output row order)
  int bb = row >> 8, s = row & 255;
  const float* src = hidden + ((size_t)s * 8 + bb) * DSZ;
  int tid = threadIdx.x;
  float vals[4]; float sum = 0.f, sq = 0.f;
  for (int i = 0; i < 4; ++i) {
    float v = src[tid + (i << 8)];
    vals[i] = v; sum += v; sq += v * v;
  }
  s1[tid] = sum; s2[tid] = sq;
  __syncthreads();
  for (int off = 128; off > 0; off >>= 1) {
    if (tid < off) { s1[tid] += s1[tid + off]; s2[tid] += s2[tid + off]; }
    __syncthreads();
  }
  float mu = s1[0] / (float)DSZ;
  float var = s2[0] / (float)DSZ - mu * mu;
  float rs = rsqrtf(var + 1e-5f);
  for (int i = 0; i < 4; ++i) {
    int d = tid + (i << 8);
    float y = (vals[i] - mu) * rs * g[d] + bt[d];
    float ge = 0.5f * y * (1.f + erff(y * 0.70710678f));
    hb[(size_t)row * DSZ + d] = f2bf(ge);
  }
}

// -------------------------------------- TDM: stage Wt tile [128n][32k] ----
__device__ __forceinline__ void tdm_issue(unsigned lds_addr,
                                          const unsigned short* wt,
                                          int n0, int kb) {
  unsigned long long ga =
      (unsigned long long)(const void*)(wt + (size_t)n0 * DSZ + kb);
  // D# group 0: count=1 | lds_addr | global_addr[56:0] | type=2
  u32x4 g0;
  g0[0] = 1u;
  g0[1] = lds_addr;
  g0[2] = (unsigned)(ga & 0xffffffffu);
  g0[3] = (unsigned)(((ga >> 32) & 0x01ffffffu) | 0x80000000u);
  // D# group 1: data_size=2B; tensor_dim0=1024, tensor_dim1=32000;
  // tile_dim0=32 (K), tile_dim1=128 (N rows); tensor_dim0_stride=1024
  i32x8 g1;
  g1[0] = 0x00010000;                         // data_size code 1 (2 bytes)
  g1[1] = (int)((DSZ & 0xffff) << 16);        // tensor_dim0 lo16 @ [63:48]
  g1[2] = (int)((DSZ >> 16) | ((VSZ & 0xffff) << 16)); // td0 hi | td1 lo
  g1[3] = (int)((VSZ >> 16) | (32 << 16));    // td1 hi | tile_dim0
  g1[4] = 128;                                // tile_dim1
  g1[5] = DSZ;                                // tensor_dim0_stride lo32
  g1[6] = 0;
  g1[7] = 0;
  i32x4 z4 = {0, 0, 0, 0};
#if defined(__clang_major__) && (__clang_major__ >= 23)
  i32x8 z8 = {0, 0, 0, 0, 0, 0, 0, 0};
  __builtin_amdgcn_tensor_load_to_lds(g0, g1, z4, z4, z8, 0);
#else
  __builtin_amdgcn_tensor_load_to_lds(g0, g1, z4, z4, 0);
#endif
}

// ------------------------------ output GEMM: [2048,1024]x[1024,32000] ------
__global__ __launch_bounds__(256)
void out_gemm(const unsigned short* __restrict__ hb,
              const unsigned short* __restrict__ wt,
              const float* __restrict__ obias, float* __restrict__ out) {
  __shared__ unsigned short btile[2][128 * 32];   // 2 x 8KB, TDM double buffer
  int lane = threadIdx.x & 31, wave = threadIdx.x >> 5;
  int n0 = blockIdx.x * 128, m0 = blockIdx.y * 64;
  int wr = wave >> 1, wc = wave & 1;              // 4 row-bands x 2 col-halves

  const unsigned short* Arow =
      hb + (size_t)(m0 + wr * 16 + (lane & 15)) * DSZ + (lane >> 4) * 16;

  v8f acc[4];
  for (int c = 0; c < 4; ++c)
    acc[c] = (v8f){0.f,0.f,0.f,0.f,0.f,0.f,0.f,0.f};

  unsigned ldsbase = (unsigned)(unsigned long long)(&btile[0][0]);
  if (wave == 0) tdm_issue(ldsbase, wt, n0, 0);
  int buf = 0;
  for (int kb = 0; kb < DSZ; kb += 32) {
    if (wave == 0) __builtin_amdgcn_s_wait_tensorcnt(0);
    __syncthreads();                               // btile[buf] ready for all
    if (wave == 0 && kb + 32 < DSZ)
      tdm_issue(ldsbase + (unsigned)((buf ^ 1) * 128 * 32 * 2), wt, n0, kb + 32);

    ABu a;
    a.u[0] = *(const v4u*)(Arow + kb);
    a.u[1] = *(const v4u*)(Arow + kb + 8);
    const unsigned short* bt = &btile[buf][0];
    for (int c = 0; c < 4; ++c) {
      int nl = wc * 64 + c * 16 + (lane & 15);     // local n row in LDS tile
      ABu b;
      b.u[0] = *(const v4u*)(bt + nl * 32 + (lane >> 4) * 16);
      b.u[1] = *(const v4u*)(bt + nl * 32 + (lane >> 4) * 16 + 8);
      acc[c] = __builtin_amdgcn_wmma_f32_16x16x32_bf16(false, a.h, false, b.h,
                                                       (short)0, acc[c],
                                                       false, false);
    }
    buf ^= 1;
  }
  // C/D layout: lane n = L&15, rows r + 8*(L>>4)
  int mbase = m0 + wr * 16 + 8 * (lane >> 4);
  for (int c = 0; c < 4; ++c) {
    int n = n0 + wc * 64 + c * 16 + (lane & 15);
    float bv = obias[n];
    for (int r = 0; r < 8; ++r)
      out[(size_t)(mbase + r) * VSZ + n] = acc[c][r] + bv;
  }
}

// -------------------------------------------------------------- launcher ---
extern "C" void kernel_launch(void* const* d_in, const int* in_sizes, int n_in,
                              void* d_out, int out_size, void* d_ws, size_t ws_size,
                              hipStream_t stream) {
  (void)in_sizes; (void)n_in; (void)out_size; (void)ws_size;
  const int*   ids     = (const int*)  d_in[0];
  const float* tok     = (const float*)d_in[1];
  const float* pos     = (const float*)d_in[2];
  const float* enc_W   = (const float*)d_in[3];
  const float* enc_b   = (const float*)d_in[4];
  const float* enc_tau = (const float*)d_in[5];
  const float* enc_bth = (const float*)d_in[6];
  const float* enc_ath = (const float*)d_in[7];
  const float* gen_W   = (const float*)d_in[8];
  const float* gen_b   = (const float*)d_in[9];
  const float* gen_tau = (const float*)d_in[10];
  const float* gen_bth = (const float*)d_in[11];
  const float* gen_ath = (const float*)d_in[12];
  const float* inf_W   = (const float*)d_in[13];
  const float* inf_b   = (const float*)d_in[14];
  const float* inf_tau = (const float*)d_in[15];
  const float* inf_bth = (const float*)d_in[16];
  const float* inf_ath = (const float*)d_in[17];
  const float* ne_g    = (const float*)d_in[18];
  const float* ne_b    = (const float*)d_in[19];
  const float* ns_g    = (const float*)d_in[20];
  const float* ns_b    = (const float*)d_in[21];
  const float* out_g   = (const float*)d_in[22];
  const float* out_bg  = (const float*)d_in[23];
  const float* out_W   = (const float*)d_in[24];
  const float* out_b   = (const float*)d_in[25];
  float* logits = (float*)d_out;

  // ---- workspace layout (all offsets 256B aligned) ----
  char* ws = (char*)d_ws;
  size_t off = 0;
  auto alloc = [&](size_t bytes) { char* p = ws + off; off += (bytes + 255) & ~(size_t)255; return p; };
  unsigned short* xbf    = (unsigned short*)alloc((size_t)SEQ * 16 * DSZ * 2);      // 8 MB
  unsigned short* wt_enc = (unsigned short*)alloc((size_t)DSZ * DSZ * 2);           // 2 MB
  unsigned short* wt_gen = (unsigned short*)alloc((size_t)LLAY * DSZ * DS2 * 2);    // 4 MB
  unsigned short* wt_inf = (unsigned short*)alloc((size_t)LLAY * DS2 * DSZ * 2);    // 4 MB
  unsigned short* wt_out = (unsigned short*)alloc((size_t)VSZ * DSZ * 2);           // 64 MB
  float*          hidden = (float*)alloc((size_t)SEQ * 8 * DSZ * 4);                // 8 MB
  unsigned short* hb     = (unsigned short*)alloc((size_t)MROW * DSZ * 2);          // 4 MB
  // zero-initialized state block (one contiguous memset)
  char* state0 = ws + off;
  float*          enc_mem   = (float*)alloc(8 * DSZ * 4);
  float*          gen_mem   = (float*)alloc(LLAY * 8 * DSZ * 4);
  float*          inf_mem   = (float*)alloc(LLAY * 8 * DS2 * 4);
  float*          states    = (float*)alloc(LLAY * 8 * DS2 * 4);
  unsigned short* states_bf = (unsigned short*)alloc(LLAY * 16 * DS2 * 2);
  float*          bu        = (float*)alloc(8 * DSZ * 4);
  unsigned short* bu_bf     = (unsigned short*)alloc(16 * DSZ * 2);
  float*          pred      = (float*)alloc(8 * DSZ * 4);
  float*          upd       = (float*)alloc(8 * DS2 * 4);
  unsigned*       bar       = (unsigned*)alloc(256);
  size_t state_bytes = (size_t)((ws + off) - state0);

  hipMemsetAsync(state0, 0, state_bytes, stream);

  // embedding -> bf16 activations (rows 8..15 zero-padded)
  embed_kernel<<<SEQ, 256, 0, stream>>>(ids, tok, pos, xbf);

  // weight transposes f32[K][N] -> bf16[N][K]
  transpose_bf16<<<dim3(DSZ/32, DSZ/32), dim3(32,8), 0, stream>>>(enc_W, wt_enc, DSZ, DSZ);
  for (int j = 0; j < LLAY; ++j) {
    transpose_bf16<<<dim3(DSZ/32, DS2/32), dim3(32,8), 0, stream>>>(
        gen_W + (size_t)j * DS2 * DSZ, wt_gen + (size_t)j * DSZ * DS2, DS2, DSZ);
    transpose_bf16<<<dim3(DS2/32, DSZ/32), dim3(32,8), 0, stream>>>(
        inf_W + (size_t)j * DSZ * DS2, wt_inf + (size_t)j * DS2 * DSZ, DSZ, DS2);
  }
  transpose_bf16<<<dim3(VSZ/32, DSZ/32), dim3(32,8), 0, stream>>>(out_W, wt_out, DSZ, VSZ);

  // persistent recurrent scan (16 WGs, global barrier per phase)
  recurrent_kernel<<<NWG_REC, 256, 0, stream>>>(
      xbf, wt_enc, wt_gen, wt_inf,
      enc_b, enc_tau, enc_bth, enc_ath,
      gen_b, gen_tau, gen_bth, gen_ath,
      inf_b, inf_tau, inf_bth, inf_ath,
      ne_g, ne_b, ns_g, ns_b,
      enc_mem, gen_mem, inf_mem, states, states_bf,
      bu, bu_bf, pred, upd, hidden, bar);

  // LN + GELU + bf16
  postln_gelu<<<MROW, 256, 0, stream>>>(hidden, out_g, out_bg, hb);

  // big projection with TDM-staged B tiles
  out_gemm<<<dim3(VSZ/128, MROW/64), 256, 0, stream>>>(hb, wt_out, out_b, logits);
}
